// StructureGraphMessagePassingInNodesV3Update_12979391168964
// MI455X (gfx1250) — matrix operand order repaired and checked by reading, not compiled, not run on hardware
//
#include <hip/hip_runtime.h>
#include <hip/hip_bf16.h>
#include <math.h>

#define Dd 1024
#define Nn 768
#define En 24576

typedef __attribute__((ext_vector_type(16))) __bf16 v16bf;
typedef __attribute__((ext_vector_type(8)))  __bf16 v8bf;
typedef __attribute__((ext_vector_type(4)))  __bf16 v4bf;
typedef __attribute__((ext_vector_type(8)))  float  v8f;

// ---------- helpers ----------
__device__ __forceinline__ unsigned f2o(float f) {
    unsigned b = __float_as_uint(f);
    return (b & 0x80000000u) ? ~b : (b | 0x80000000u);
}
__device__ __forceinline__ float o2f(unsigned u) {
    unsigned b = (u & 0x80000000u) ? (u ^ 0x80000000u) : ~u;
    return __uint_as_float(b);
}
__device__ __forceinline__ v16bf cat8(v8bf a, v8bf b) {
    return __builtin_shufflevector(a, b, 0, 1, 2, 3, 4, 5, 6, 7,
                                         8, 9, 10, 11, 12, 13, 14, 15);
}

// ---------- generic split-bf16 WMMA GEMM: C[M][N] = A[M][K] @ B[K][N] ----------
// block tile 32(M) x 64(N), 8 waves, 1 16x16 fragment each; A row stride == K.
__global__ __launch_bounds__(256) void gemm_kernel(const float* __restrict__ A,
                                                   const float* __restrict__ B,
                                                   float* __restrict__ C,
                                                   int K, int ldb, int ldc) {
    __shared__ float Apan[32][33];
    const int t    = threadIdx.x;
    const int lane = t & 31, w = t >> 5;
    const int er   = w >> 2;          // 0..1 row strip
    const int nh   = w & 3;           // 0..3 col strip (16 cols)
    const int m    = lane & 15, half = lane >> 4;
    const int rowbase = blockIdx.x * 32;
    const int colbase = blockIdx.y * 64 + nh * 16;
    const int e = t >> 3, kq = (t & 7) * 4;   // cooperative panel load mapping
    v8f acc = {};
    for (int kb = 0; kb < K; kb += 32) {
        __syncthreads();
        const float4 v = *(const float4*)(A + (size_t)(rowbase + e) * K + kb + kq);
        Apan[e][kq + 0] = v.x; Apan[e][kq + 1] = v.y;
        Apan[e][kq + 2] = v.z; Apan[e][kq + 3] = v.w;
        if (kb + 32 < K)
            __builtin_prefetch(A + (size_t)(rowbase + e) * K + kb + 32 + kq, 0, 0);
        __syncthreads();
        v16bf ah, al;
#pragma unroll
        for (int i = 0; i < 16; ++i) {
            int k = (i & 7) + 8 * (2 * (i >> 3) + half);     // 16-bit A frag layout
            float x = Apan[er * 16 + m][k];
            __bf16 h = (__bf16)x; ah[i] = h; al[i] = (__bf16)(x - (float)h);
        }
        v16bf bh, bl;
        const int col = colbase + m;
#pragma unroll
        for (int i = 0; i < 16; ++i) {
            int k = kb + (i & 7) + 8 * (2 * (i >> 3) + half); // 16-bit B frag layout
            float x = B[(size_t)k * ldb + col];
            __bf16 h = (__bf16)x; bh[i] = h; bl[i] = (__bf16)(x - (float)h);
        }
        acc = __builtin_amdgcn_wmma_f32_16x16x32_bf16(false, ah, false, bh, (short)0, acc, false, false);
        acc = __builtin_amdgcn_wmma_f32_16x16x32_bf16(false, ah, false, bl, (short)0, acc, false, false);
        acc = __builtin_amdgcn_wmma_f32_16x16x32_bf16(false, al, false, bh, (short)0, acc, false, false);
    }
#pragma unroll
    for (int r = 0; r < 8; ++r)
        C[(size_t)(rowbase + er * 16 + r + 8 * half) * ldc + colbase + m] = acc[r];
}

// ---------- transpose + hi/lo split of a 1024x1024 f32 matrix ----------
// X[K][N] f32  ->  Thi/Tlo[N][K] bf16 (K contiguous per output row)
__global__ __launch_bounds__(256) void splitT_kernel(const float* __restrict__ X,
                                                     __bf16* __restrict__ Thi,
                                                     __bf16* __restrict__ Tlo) {
    __shared__ float tile[32][33];
    const int k0  = blockIdx.x * 32;
    const int n0  = blockIdx.y * 32;
    const int tx  = threadIdx.x & 31;
    const int ty0 = (threadIdx.x >> 5) * 4;
#pragma unroll
    for (int j = 0; j < 4; ++j)
        tile[ty0 + j][tx] = X[(size_t)(k0 + ty0 + j) * Dd + n0 + tx];
    __syncthreads();
#pragma unroll
    for (int j = 0; j < 4; ++j) {
        float x = tile[tx][ty0 + j];
        __bf16 h = (__bf16)x;
        Thi[(size_t)(n0 + ty0 + j) * Dd + k0 + tx] = h;
        Tlo[(size_t)(n0 + ty0 + j) * Dd + k0 + tx] = (__bf16)(x - (float)h);
    }
}

// ---------- fused edge-logits kernel ----------
// Per block: 32 edges. Computes Gs = R@Ks, Go = R@Ko tile-by-tile (64 cols at a
// time), adds gathered node terms + biases, accumulates ts.to, emits logits only.
// A is split into bf16 hi/lo once at LDS staging; B comes from pre-split
// transposed planes, so the K-loop has no float->bf16 conversion math at all.
__global__ __launch_bounds__(256) void edge_logits_kernel(
        const float* __restrict__ R,
        const __bf16* __restrict__ KsThi, const __bf16* __restrict__ KsTlo,
        const __bf16* __restrict__ KoThi, const __bf16* __restrict__ KoTlo,
        const float* __restrict__ Ts, const float* __restrict__ Us,
        const float* __restrict__ To, const float* __restrict__ Uo,
        const float* __restrict__ bs, const float* __restrict__ bo,
        const int* __restrict__ sbj, const int* __restrict__ obj,
        float* __restrict__ logits) {
    __shared__ __bf16 Ahi[32 * 40];          // 32 edges x 32 k, stride 40 (bank-spread, 16B-aligned rows)
    __shared__ __bf16 Alo[32 * 40];
    __shared__ float  Gbuf[2][32][65];
    __shared__ int    sed[32], oed[32];
    const int t    = threadIdx.x;
    const int lane = t & 31, w = t >> 5;
    const int mm   = w & 1;            // 0 = Gs-side, 1 = Go-side
    const int er   = (w >> 1) & 1;     // edge strip (16 edges)
    const int nh   = w >> 2;           // 0..1 (32 cols each, 2 frags)
    const int m    = lane & 15, half = lane >> 4;
    const int eb   = blockIdx.x * 32;
    const int e = t >> 3, kq = (t & 7) * 4;
    const int ee = t & 31;             // epilogue: edge handled by this thread
    const int cc = (t >> 5) * 8;       // epilogue: 8-col slice
    if (t < 32) { sed[t] = sbj[eb + t]; oed[t] = obj[eb + t]; }
    __syncthreads();
    const int s_e = sed[ee], o_e = oed[ee];
    const __bf16* BThi = mm ? KoThi : KsThi;
    const __bf16* BTlo = mm ? KoTlo : KsTlo;
    float acc_log = 0.0f;
    for (int nc = 0; nc < 16; ++nc) {
        const int ncb = nc * 64;
        v8f acc0 = {}, acc1 = {};
        for (int kb = 0; kb < Dd; kb += 32) {
            __syncthreads();
            const float4 v = *(const float4*)(R + (size_t)(eb + e) * Dd + kb + kq);
            v4bf hh, ll;
            {
                __bf16 h0 = (__bf16)v.x; hh[0] = h0; ll[0] = (__bf16)(v.x - (float)h0);
                __bf16 h1 = (__bf16)v.y; hh[1] = h1; ll[1] = (__bf16)(v.y - (float)h1);
                __bf16 h2 = (__bf16)v.z; hh[2] = h2; ll[2] = (__bf16)(v.z - (float)h2);
                __bf16 h3 = (__bf16)v.w; hh[3] = h3; ll[3] = (__bf16)(v.w - (float)h3);
            }
            *(v4bf*)(Ahi + e * 40 + kq) = hh;
            *(v4bf*)(Alo + e * 40 + kq) = ll;
            if (kb + 32 < Dd)
                __builtin_prefetch(R + (size_t)(eb + e) * Dd + kb + 32 + kq, 0, 0);
            __syncthreads();
            // A fragments: pure LDS b128 loads, no conversion
            const __bf16* arow = Ahi + (er * 16 + m) * 40;
            const __bf16* brow = Alo + (er * 16 + m) * 40;
            v16bf ah = cat8(*(const v8bf*)(arow + 8 * half),
                            *(const v8bf*)(arow + 16 + 8 * half));
            v16bf al = cat8(*(const v8bf*)(brow + 8 * half),
                            *(const v8bf*)(brow + 16 + 8 * half));
            // B fragments: contiguous bf16 from transposed planes
            const int col0 = ncb + nh * 32 + m;
            const __bf16* bh0p = BThi + (size_t)col0 * Dd + kb + 8 * half;
            const __bf16* bl0p = BTlo + (size_t)col0 * Dd + kb + 8 * half;
            const __bf16* bh1p = bh0p + (size_t)16 * Dd;
            const __bf16* bl1p = bl0p + (size_t)16 * Dd;
            v16bf bh0 = cat8(*(const v8bf*)(bh0p), *(const v8bf*)(bh0p + 16));
            v16bf bl0 = cat8(*(const v8bf*)(bl0p), *(const v8bf*)(bl0p + 16));
            v16bf bh1 = cat8(*(const v8bf*)(bh1p), *(const v8bf*)(bh1p + 16));
            v16bf bl1 = cat8(*(const v8bf*)(bl1p), *(const v8bf*)(bl1p + 16));
            acc0 = __builtin_amdgcn_wmma_f32_16x16x32_bf16(false, ah, false, bh0, (short)0, acc0, false, false);
            acc0 = __builtin_amdgcn_wmma_f32_16x16x32_bf16(false, ah, false, bl0, (short)0, acc0, false, false);
            acc0 = __builtin_amdgcn_wmma_f32_16x16x32_bf16(false, al, false, bh0, (short)0, acc0, false, false);
            acc1 = __builtin_amdgcn_wmma_f32_16x16x32_bf16(false, ah, false, bh1, (short)0, acc1, false, false);
            acc1 = __builtin_amdgcn_wmma_f32_16x16x32_bf16(false, ah, false, bl1, (short)0, acc1, false, false);
            acc1 = __builtin_amdgcn_wmma_f32_16x16x32_bf16(false, al, false, bh1, (short)0, acc1, false, false);
        }
        __syncthreads();
#pragma unroll
        for (int r = 0; r < 8; ++r) {
            int ge = er * 16 + r + 8 * half;
            Gbuf[mm][ge][nh * 32 + m]      = acc0[r];
            Gbuf[mm][ge][nh * 32 + 16 + m] = acc1[r];
        }
        __syncthreads();
#pragma unroll
        for (int j = 0; j < 8; ++j) {
            int col = ncb + cc + j;
            float tsv = Gbuf[0][ee][cc + j] + Ts[(size_t)s_e * Dd + col]
                      + Us[(size_t)o_e * Dd + col] + bs[col];
            float tov = Gbuf[1][ee][cc + j] + To[(size_t)o_e * Dd + col]
                      + Uo[(size_t)s_e * Dd + col] + bo[col];
            acc_log += tsv * tov;
        }
    }
    __syncthreads();
    float* red = &Gbuf[0][0][0];
    red[ee * 8 + (t >> 5)] = acc_log;
    __syncthreads();
    if (t < 32) {
        float s = 0.f;
#pragma unroll
        for (int j = 0; j < 8; ++j) s += red[t * 8 + j];
        logits[eb + t] = s * 0.03125f;   // 1/sqrt(1024)
    }
}

// ---------- small kernels ----------
__global__ void init_kernel(float* ctx, unsigned* encS, unsigned* encO,
                            float* sumS, float* sumO) {
    int i = blockIdx.x * blockDim.x + threadIdx.x;
    if (i < Nn * Dd) ctx[i] = 0.f;
    if (i < Nn) { encS[i] = 0u; encO[i] = 0u; sumS[i] = 0.f; sumO[i] = 0.f; }
}

__global__ void addw_kernel(const float* Wsv, const float* P1a, float* Ms,
                            const float* Wov, const float* P2b, float* Mo) {
    int i = blockIdx.x * blockDim.x + threadIdx.x;
    if (i < Dd * Dd) { Ms[i] = Wsv[i] + P1a[i]; Mo[i] = Wov[i] + P2b[i]; }
}

__global__ void bias_kernel(const float* brel, const float* Wsr, const float* Wor,
                            const float* bsbj, const float* bobj,
                            float* bs, float* bo) {
    int j = blockIdx.x * blockDim.x + threadIdx.x;
    if (j >= Dd) return;
    float as = 0.f, ao = 0.f;
    for (int k = 0; k < Dd; ++k) {
        float br = brel[k];
        as += br * Wsr[(size_t)k * Dd + j];
        ao += br * Wor[(size_t)k * Dd + j];
    }
    bs[j] = bsbj[j] + as;
    bo[j] = bobj[j] + ao;
}

__global__ void max_kernel(const float* logits, const int* sbj, const int* obj,
                           unsigned* encS, unsigned* encO) {
    int e = blockIdx.x * blockDim.x + threadIdx.x;
    if (e >= En) return;
    unsigned v = f2o(logits[e]);
    atomicMax(&encS[sbj[e]], v);
    atomicMax(&encO[obj[e]], v);
}

__global__ void expsum_kernel(const float* logits, const int* sbj, const int* obj,
                              const unsigned* encS, const unsigned* encO,
                              float* es, float* eo, float* sumS, float* sumO) {
    int e = blockIdx.x * blockDim.x + threadIdx.x;
    if (e >= En) return;
    float l = logits[e];
    int s = sbj[e], o = obj[e];
    float x = expf(l - o2f(encS[s]));
    float y = expf(l - o2f(encO[o]));
    es[e] = x; eo[e] = y;
    atomicAdd(&sumS[s], x);
    atomicAdd(&sumO[o], y);
}

__global__ __launch_bounds__(256) void scatter_kernel(
        const float* __restrict__ vf, const int* __restrict__ sbj,
        const int* __restrict__ obj, const float* __restrict__ es,
        const float* __restrict__ eo, const float* __restrict__ sumS,
        const float* __restrict__ sumO, float* __restrict__ ctx) {
    int e = blockIdx.x;
    int c = threadIdx.x * 4;
    int s = sbj[e], o = obj[e];
    float ws = es[e] / sumS[s];
    float wo = eo[e] / sumO[o];
    float4 vo4 = *(const float4*)(vf + (size_t)o * Dd + c);
    float4 vs4 = *(const float4*)(vf + (size_t)s * Dd + c);
    atomicAdd(&ctx[(size_t)s * Dd + c + 0], ws * vo4.x);
    atomicAdd(&ctx[(size_t)s * Dd + c + 1], ws * vo4.y);
    atomicAdd(&ctx[(size_t)s * Dd + c + 2], ws * vo4.z);
    atomicAdd(&ctx[(size_t)s * Dd + c + 3], ws * vo4.w);
    atomicAdd(&ctx[(size_t)o * Dd + c + 0], wo * vs4.x);
    atomicAdd(&ctx[(size_t)o * Dd + c + 1], wo * vs4.y);
    atomicAdd(&ctx[(size_t)o * Dd + c + 2], wo * vs4.z);
    atomicAdd(&ctx[(size_t)o * Dd + c + 3], wo * vs4.w);
}

__global__ void update_kernel(const float* vf, const float* ctxW, const float* bctx,
                              const unsigned* encS, const unsigned* encO, float* out) {
    int i = blockIdx.x * blockDim.x + threadIdx.x;
    if (i >= Nn * Dd) return;
    int n = i >> 10, c = i & 1023;
    bool inv = (encS[n] | encO[n]) != 0u;   // node has >=1 incident edge
    out[i] = vf[i] + (inv ? (ctxW[i] + bctx[c]) : 0.0f);
}

// ---------- launcher ----------
extern "C" void kernel_launch(void* const* d_in, const int* in_sizes, int n_in,
                              void* d_out, int out_size, void* d_ws, size_t ws_size,
                              hipStream_t stream) {
    (void)in_sizes; (void)n_in; (void)out_size; (void)ws_size;
    const float* vf   = (const float*)d_in[0];
    const float* R    = (const float*)d_in[1];
    const float* Wrel = (const float*)d_in[2];
    const float* brel = (const float*)d_in[3];
    const float* Wsbj = (const float*)d_in[4];
    const float* bsbj = (const float*)d_in[5];
    const float* Wobj = (const float*)d_in[6];
    const float* bobj = (const float*)d_in[7];
    const float* Wctx = (const float*)d_in[8];
    const float* bctx = (const float*)d_in[9];
    const int*   sbj  = (const int*)d_in[10];
    const int*   obj  = (const int*)d_in[11];
    float* out = (float*)d_out;

    const size_t DD = (size_t)Dd * Dd;
    float* p  = (float*)d_ws;
    float* P1 = p; p += 3 * DD;          // W_rel @ Ws_r  (rows: Ms-part | Ns | Ks)
    float* P2 = p; p += 3 * DD;          // W_rel @ Wo_r  (rows: No | Mo-part | Ko)
    float* Ms = p; p += DD;
    float* Mo = p; p += DD;
    float* Ts = p; p += (size_t)Nn * Dd;
    float* Us = p; p += (size_t)Nn * Dd;
    float* To = p; p += (size_t)Nn * Dd;
    float* Uo = p; p += (size_t)Nn * Dd;
    float* bs = p; p += Dd;
    float* bo = p; p += Dd;
    float* logits = p; p += En;
    float* es = p; p += En;
    float* eo = p; p += En;
    float* sumS = p; p += Nn;
    float* sumO = p; p += Nn;
    unsigned* encS = (unsigned*)p; p += Nn;
    unsigned* encO = (unsigned*)p; p += Nn;
    float* ctx  = p; p += (size_t)Nn * Dd;
    float* ctxW = p; p += (size_t)Nn * Dd;
    // bf16 transposed planes for the fused kernel's B operands (each D*D bf16 = DD/2 floats)
    __bf16* KsThi = (__bf16*)p; p += DD / 2;
    __bf16* KsTlo = (__bf16*)p; p += DD / 2;
    __bf16* KoThi = (__bf16*)p; p += DD / 2;
    __bf16* KoTlo = (__bf16*)p; p += DD / 2;

    const float* Wsr = Wsbj + DD;   // Ws_r
    const float* Wor = Wobj + DD;   // Wo_r

    // Output 0: rel_visual_feat passthrough
    hipMemcpyAsync(out, R, (size_t)En * Dd * sizeof(float),
                   hipMemcpyDeviceToDevice, stream);

    init_kernel<<<(Nn * Dd + 255) / 256, 256, 0, stream>>>(ctx, encS, encO, sumS, sumO);

    // Weight-combine GEMMs (3072 x 1024 x 1024 each)
    gemm_kernel<<<dim3(3 * Dd / 32, Dd / 64), 256, 0, stream>>>(Wrel, Wsr, P1, Dd, Dd, Dd);
    gemm_kernel<<<dim3(3 * Dd / 32, Dd / 64), 256, 0, stream>>>(Wrel, Wor, P2, Dd, Dd, Dd);
    addw_kernel<<<(DD + 255) / 256, 256, 0, stream>>>(Wsbj, P1, Ms, Wobj, P2 + DD, Mo);
    bias_kernel<<<(Dd + 255) / 256, 256, 0, stream>>>(brel, Wsr, Wor, bsbj, bobj, bs, bo);

    // Pre-split + transpose the fused kernel's B matrices (Ks, Ko)
    splitT_kernel<<<dim3(Dd / 32, Dd / 32), 256, 0, stream>>>(P1 + 2 * DD, KsThi, KsTlo);
    splitT_kernel<<<dim3(Dd / 32, Dd / 32), 256, 0, stream>>>(P2 + 2 * DD, KoThi, KoTlo);

    // Node GEMMs (768 x 1024 x 1024)
    gemm_kernel<<<dim3(Nn / 32, Dd / 64), 256, 0, stream>>>(vf, Ms,      Ts, Dd, Dd, Dd);
    gemm_kernel<<<dim3(Nn / 32, Dd / 64), 256, 0, stream>>>(vf, P1 + DD, Us, Dd, Dd, Dd);
    gemm_kernel<<<dim3(Nn / 32, Dd / 64), 256, 0, stream>>>(vf, Mo,      To, Dd, Dd, Dd);
    gemm_kernel<<<dim3(Nn / 32, Dd / 64), 256, 0, stream>>>(vf, P2,      Uo, Dd, Dd, Dd);

    // Dominant fused edge GEMM + logits (2 x 24576 x 1024 x 1024, never materialized)
    edge_logits_kernel<<<En / 32, 256, 0, stream>>>(R, KsThi, KsTlo, KoThi, KoTlo,
                                                    Ts, Us, To, Uo, bs, bo,
                                                    sbj, obj, logits);

    // Dual edge-softmax + weighted scatter
    max_kernel<<<(En + 255) / 256, 256, 0, stream>>>(logits, sbj, obj, encS, encO);
    expsum_kernel<<<(En + 255) / 256, 256, 0, stream>>>(logits, sbj, obj, encS, encO,
                                                        es, eo, sumS, sumO);
    scatter_kernel<<<En, 256, 0, stream>>>(vf, sbj, obj, es, eo, sumS, sumO, ctx);

    // ctx @ W_ctx and masked update
    gemm_kernel<<<dim3(Nn / 32, Dd / 64), 256, 0, stream>>>(ctx, Wctx, ctxW, Dd, Dd, Dd);
    update_kernel<<<(Nn * Dd + 255) / 256, 256, 0, stream>>>(vf, ctxW, bctx, encS, encO,
                                                             out + (size_t)En * Dd);
}
